// GlobalRouters_21157008900534
// MI455X (gfx1250) — compile-verified, matrix operand order checked by reading, not run on hardware
//
#include <hip/hip_runtime.h>
#include <hip/hip_bf16.h>

typedef __attribute__((ext_vector_type(16))) __bf16 v16bf;
typedef __attribute__((ext_vector_type(8)))  __bf16 v8bf;
typedef __attribute__((ext_vector_type(8)))  float  v8f;

#define B_SZ   16
#define S_SZ   2048
#define D_SZ   1024
#define NCOL   320            // 128 compress + 64 Q + 64 K + 64 V
#define ROWS   32             // tokens per workgroup (2 WMMA row-tiles)
#define XSTR   (D_SZ + 8)     // padded bf16 row stride (2064B) -> conflict-free ds_load_b128
#define LSTR   (NCOL + 4)     // padded f32 logit row stride
#define EPS_F  1e-8f

#define XBYTES (ROWS * XSTR * 2)   // 66048
#define LBYTES (ROWS * LSTR * 4)   // 41472 (< XBYTES, unioned)

// ---------------------------------------------------------------------------
// Kernel 0: convert concatenated router weights f32 -> bf16; zero accumulators
// ---------------------------------------------------------------------------
__global__ __launch_bounds__(256) void gr_prep(const float* __restrict__ Wc,
                                               const float* __restrict__ Wq,
                                               const float* __restrict__ Wk,
                                               const float* __restrict__ Wv,
                                               __bf16* __restrict__ Wb,
                                               float* __restrict__ wun) {
    int i = blockIdx.x * 256 + threadIdx.x;
    if (i < B_SZ * NCOL) wun[i] = 0.0f;
    if (i >= NCOL * D_SZ) return;
    int row = i >> 10;
    int col = i & (D_SZ - 1);
    float v;
    if (row < 128)      v = Wc[row * D_SZ + col];
    else if (row < 192) v = Wq[(row - 128) * D_SZ + col];
    else if (row < 256) v = Wk[(row - 192) * D_SZ + col];
    else                v = Wv[(row - 256) * D_SZ + col];
    Wb[i] = (__bf16)v;
}

// ---------------------------------------------------------------------------
// Kernel 1: fused 4-router GEMM (bf16 WMMA, ping-pong pipelined) + softmax +
// importance-weighted reduction. Workgroup = 32 tokens of one batch.
// 5 waves x (2 row-tiles x 4 col-tiles) = 32 rows x 320 cols.
// x-tile and logit buffers are unioned: ~67 KB LDS -> 4 workgroups per WGP.
// ---------------------------------------------------------------------------
__global__ __launch_bounds__(160) void gr_main(const float* __restrict__ x,
                                               const float* __restrict__ imp,
                                               const __bf16* __restrict__ Wb,
                                               float* __restrict__ wun) {
    __shared__ __align__(16) char smem[XBYTES];        // xt (GEMM phase) / logit (reduce phase)
    __shared__ float impr[ROWS];
    __shared__ float rmax[4][ROWS];
    __shared__ float rinv[4][ROWS];

    __bf16* xt    = (__bf16*)smem;
    float*  logit = (float*)smem;

    const int tid = threadIdx.x;
    const int b   = blockIdx.y;
    const long long s0 = (long long)blockIdx.x * ROWS;
    const float* xbase = x + ((long long)b * S_SZ + s0) * D_SZ;

    // ---- stage x tile: f32 global -> bf16 LDS (float4 loads) ----
    for (int i = tid; i < ROWS * (D_SZ / 4); i += 160) {
        int row = i / (D_SZ / 4);
        int c4  = (i - row * (D_SZ / 4)) * 4;
        float4 v = *(const float4*)(xbase + (long long)row * D_SZ + c4);
        __bf16* dst = &xt[row * XSTR + c4];
        dst[0] = (__bf16)v.x; dst[1] = (__bf16)v.y;
        dst[2] = (__bf16)v.z; dst[3] = (__bf16)v.w;
    }
    if (tid < ROWS) impr[tid] = imp[(long long)b * S_SZ + s0 + tid];
    __syncthreads();

    // ---- WMMA GEMM: each wave owns 64 columns (4 tiles) x 32 rows (2 tiles) ----
    const int wave  = tid >> 5;
    const int lane  = tid & 31;
    const int lhalf = lane >> 4;        // ISA A/B layout: K-split by lane half
    const int lmod  = lane & 15;        // A: row M ; B: column N
    const int colbase = wave * 64;
    const int koff    = lhalf * 8;      // bf16 offset inside 32-wide K chunk

    v8f acc[2][4] = {};
    v16bf aP[2], aQ[2], bP[4], bQ[4];   // ping-pong buffers (no copies)

#define LOAD_A(dst, kb)                                                         \
    do {                                                                        \
        _Pragma("unroll")                                                       \
        for (int rt = 0; rt < 2; ++rt) {                                        \
            const __bf16* ap = &xt[(rt * 16 + lmod) * XSTR + (kb) + koff];      \
            v8bf a0 = *(const v8bf*)(ap);                                       \
            v8bf a1 = *(const v8bf*)(ap + 16);                                  \
            dst[rt] = __builtin_shufflevector(a0, a1, 0, 1, 2, 3, 4, 5, 6, 7,   \
                                              8, 9, 10, 11, 12, 13, 14, 15);    \
        }                                                                       \
    } while (0)

#define LOAD_B(dst, kb)                                                         \
    do {                                                                        \
        _Pragma("unroll")                                                       \
        for (int t = 0; t < 4; ++t) {                                           \
            const __bf16* bp = Wb + (long long)(colbase + t * 16 + lmod) * D_SZ \
                                  + (kb) + koff;                                \
            v8bf b0 = *(const v8bf*)(bp);                                       \
            v8bf b1 = *(const v8bf*)(bp + 16);                                  \
            dst[t] = __builtin_shufflevector(b0, b1, 0, 1, 2, 3, 4, 5, 6, 7,    \
                                             8, 9, 10, 11, 12, 13, 14, 15);     \
        }                                                                       \
    } while (0)

#define WMMA_STEP(A, Bv)                                                        \
    do {                                                                        \
        _Pragma("unroll")                                                       \
        for (int rt = 0; rt < 2; ++rt)                                          \
            _Pragma("unroll")                                                   \
            for (int t = 0; t < 4; ++t)                                         \
                acc[rt][t] = __builtin_amdgcn_wmma_f32_16x16x32_bf16(           \
                    false, A[rt], false, Bv[t], (short)0, acc[rt][t],           \
                    false, false);                                              \
    } while (0)

    LOAD_A(aP, 0);
    LOAD_B(bP, 0);

    for (int kb = 0; kb < D_SZ; kb += 64) {
        // half-step 1: prefetch kb+32 into Q, compute on P
        LOAD_A(aQ, kb + 32);
        LOAD_B(bQ, kb + 32);
        WMMA_STEP(aP, bP);
        // half-step 2: prefetch kb+64 into P (wrap keeps loads in-bounds), compute on Q
        const int kn = (kb + 64 < D_SZ) ? (kb + 64) : 0;
        LOAD_A(aP, kn);
        LOAD_B(bP, kn);
        WMMA_STEP(aQ, bQ);
    }
#undef LOAD_A
#undef LOAD_B
#undef WMMA_STEP

    // all waves done reading xt before logits overwrite the same LDS
    __syncthreads();

    // ---- scatter C fragments to LDS logits (M = rt*16 + i + lhalf*8, N) ----
#pragma unroll
    for (int rt = 0; rt < 2; ++rt)
#pragma unroll
        for (int t = 0; t < 4; ++t) {
            const int n = colbase + t * 16 + lmod;
#pragma unroll
            for (int i = 0; i < 8; ++i)
                logit[(rt * 16 + i + lhalf * 8) * LSTR + n] = acc[rt][t][i];
        }
    __syncthreads();

    // ---- per-(row, router) softmax stats: 32 rows x 4 routers = 128 tasks ----
    if (tid < 128) {
        const int row = tid & 31;
        const int r   = tid >> 5;
        const int cb  = (r == 0) ? 0 : 128 + (r - 1) * 64;
        const int nc  = (r == 0) ? 128 : 64;
        float mx = -__builtin_inff();
        for (int c = 0; c < nc; ++c) mx = fmaxf(mx, logit[row * LSTR + cb + c]);
        float s = 0.0f;
        for (int c = 0; c < nc; ++c) s += __expf(logit[row * LSTR + cb + c] - mx);
        rmax[r][row] = mx;
        rinv[r][row] = 1.0f / s;
    }
    __syncthreads();

    // ---- importance-weighted column sums -> global accumulator ----
    for (int col = tid; col < NCOL; col += 160) {
        const int r = (col < 128) ? 0 : (col < 192) ? 1 : (col < 256) ? 2 : 3;
        float a = 0.0f;
#pragma unroll 8
        for (int row = 0; row < ROWS; ++row) {
            float p = __expf(logit[row * LSTR + col] - rmax[r][row]) * rinv[r][row];
            a += impr[row] * p;
        }
        atomicAdd(&wun[b * NCOL + col], a);
    }
}

// ---------------------------------------------------------------------------
// Kernel 2: normalize, top-k, renormalize, emit (weights + indices-as-float)
// ---------------------------------------------------------------------------
__global__ __launch_bounds__(64) void gr_finalize(const float* __restrict__ wun,
                                                  float* __restrict__ out) {
    const int t = threadIdx.x;
    const int b = t >> 2;
    const int r = t & 3;
    const int cb[4]  = {0, 128, 192, 256};
    const int nc[4]  = {128, 64, 64, 64};
    const int kk[4]  = {16, 8, 8, 8};
    const int ow[4]  = {0, 512, 768, 1024};     // weight output bases
    const int oi[4]  = {256, 640, 896, 1152};   // index  output bases
    const int n = nc[r], k = kk[r];

    float v[128];
    float s = 0.0f;
    for (int i = 0; i < n; ++i) {
        float x = wun[b * NCOL + cb[r] + i];
        v[i] = x;
        s += x;
    }
    const float inv = 1.0f / (s + EPS_F);
    for (int i = 0; i < n; ++i) v[i] *= inv;

    float tv[16];
    int   ti[16];
    for (int j = 0; j < k; ++j) {               // strict '>' keeps first index (lax.top_k tie-break)
        float best = -__builtin_inff();
        int bi = 0;
        for (int i = 0; i < n; ++i)
            if (v[i] > best) { best = v[i]; bi = i; }
        tv[j] = best; ti[j] = bi;
        v[bi] = -__builtin_inff();
    }
    float ts = 0.0f;
    for (int j = 0; j < k; ++j) ts += tv[j];
    const float tinv = 1.0f / (ts + EPS_F);
    for (int j = 0; j < k; ++j) {
        out[ow[r] + b * k + j] = tv[j] * tinv;
        out[oi[r] + b * k + j] = (float)ti[j];
    }
}

// ---------------------------------------------------------------------------
extern "C" void kernel_launch(void* const* d_in, const int* in_sizes, int n_in,
                              void* d_out, int out_size, void* d_ws, size_t ws_size,
                              hipStream_t stream) {
    const float* x   = (const float*)d_in[0];
    const float* imp = (const float*)d_in[1];
    const float* Wc  = (const float*)d_in[2];
    const float* Wq  = (const float*)d_in[3];
    const float* Wk  = (const float*)d_in[4];
    const float* Wv  = (const float*)d_in[5];

    __bf16* Wb  = (__bf16*)d_ws;                                        // 320*1024 bf16
    float*  wun = (float*)((char*)d_ws + NCOL * D_SZ * sizeof(__bf16)); // 16*320 f32

    gr_prep<<<dim3((NCOL * D_SZ + 255) / 256), dim3(256), 0, stream>>>(Wc, Wq, Wk, Wv, Wb, wun);
    gr_main<<<dim3(S_SZ / ROWS, B_SZ), dim3(160), 0, stream>>>(x, imp, Wb, wun);
    gr_finalize<<<dim3(1), dim3(64), 0, stream>>>(wun, (float*)d_out);
}
